// MultiAttentionHead_72155450572891
// MI455X (gfx1250) — matrix operand order repaired and checked by reading, not compiled, not run on hardware
//
#include <hip/hip_runtime.h>

// ---------------- problem constants (match reference) ----------------
constexpr int Bc  = 2;
constexpr int Sc  = 2048;
constexpr int Dc  = 1024;
constexpr int Hc  = 16;
constexpr int HDc = 64;                 // head dim
constexpr float NEGV = -9000000000000000.0f;

// ---------------- detect CDNA5 async-to-LDS path ----------------
#if defined(__has_builtin)
#if __has_builtin(__builtin_amdgcn_global_load_async_to_lds_b128) && \
    __has_builtin(__builtin_amdgcn_s_wait_asynccnt)
#define USE_ASYNC_LDS 1
#endif
#endif
#ifndef USE_ASYNC_LDS
#define USE_ASYNC_LDS 0
#endif

// ---------------- WMMA types ----------------
typedef __attribute__((ext_vector_type(16))) __bf16          v16bf;
typedef __attribute__((ext_vector_type(16))) unsigned short  v16u;
typedef __attribute__((ext_vector_type(8)))  unsigned short  u16x8;
typedef __attribute__((ext_vector_type(4)))  unsigned short  u16x4;
typedef __attribute__((ext_vector_type(8)))  float           v8f;

union FragCvt { v16u u; v16bf b; };

__device__ __forceinline__ unsigned short f2bf(float f) {
  unsigned u = __float_as_uint(f);
  u += 0x7FFFu + ((u >> 16) & 1u);      // round-to-nearest-even
  return (unsigned short)(u >> 16);
}

__device__ __forceinline__ u16x4 pack4(float4 w) {
  u16x4 r = {f2bf(w.x), f2bf(w.y), f2bf(w.z), f2bf(w.w)};
  return r;
}

// build a 16-element bf16 fragment from two contiguous 8x16b chunks
__device__ __forceinline__ v16bf make_frag(const unsigned short* p0,
                                           const unsigned short* p1) {
  u16x8 lo = *(const u16x8*)p0;
  u16x8 hi = *(const u16x8*)p1;
  FragCvt f;
#pragma unroll
  for (int i = 0; i < 8; ++i) { f.u[i] = lo[i]; f.u[i + 8] = hi[i]; }
  return f.b;
}

__device__ __forceinline__ v8f wmma_bf16(v16bf a, v16bf b, v8f c) {
  return __builtin_amdgcn_wmma_f32_16x16x32_bf16(
      false, a, false, b, (short)0, c, false, false);
}

#if USE_ASYNC_LDS
// Builtin parameter type discovered from compiler diagnostic:
//   v4i (vector_size 16 of int) pointers in addrspace(1) / addrspace(3)
typedef int v4i_vs __attribute__((vector_size(16)));
typedef __attribute__((address_space(1))) v4i_vs* gptr_v4i;
typedef __attribute__((address_space(3))) v4i_vs* lptr_v4i;

// 16-byte async DMA: global -> LDS, tracked by ASYNCcnt
__device__ __forceinline__ void async_copy16(const unsigned short* g,
                                             unsigned short* l) {
  unsigned short* gnc = const_cast<unsigned short*>(g);
  __builtin_amdgcn_global_load_async_to_lds_b128(
      (gptr_v4i)(gnc), (lptr_v4i)(l), 0, 0);
}
#endif

// =====================================================================
// Kernel 1: fused QKV projection.  grid = (M/16, D/128, 3), block = 256
// =====================================================================
__global__ __launch_bounds__(256) void qkv_proj_kernel(
    const float* __restrict__ k_in, const float* __restrict__ q_in,
    const float* __restrict__ v_in,
    const float* __restrict__ Wq, const float* __restrict__ bq,
    const float* __restrict__ Wk, const float* __restrict__ bk,
    const float* __restrict__ Wv, const float* __restrict__ bv,
    unsigned short* __restrict__ Qh, unsigned short* __restrict__ Kh,
    unsigned short* __restrict__ Vh) {
  const int which = blockIdx.z;
  const float* X    = (which == 0) ? q_in : (which == 1) ? k_in : v_in;
  const float* W    = (which == 0) ? Wq   : (which == 1) ? Wk   : Wv;
  const float* bias = (which == 0) ? bq   : (which == 1) ? bk   : bv;
  unsigned short* Out = (which == 0) ? Qh : (which == 1) ? Kh : Vh;

  __shared__ unsigned short As[16][32];    // A tile  [m][k]
  __shared__ unsigned short Bs[128][32];   // W tile transposed [n][k]

  const int tid  = threadIdx.x;
  const int lane = tid & 31, wv = tid >> 5;
  const int rowbase = blockIdx.x * 16;
  const int col0    = blockIdx.y * 128;
  const int kbA = (lane >> 4) * 16, kbB = (lane >> 4) * 16;

  v8f acc = {};
  for (int kt = 0; kt < Dc / 32; ++kt) {
    const int kk0 = kt * 32;
    if (tid < 128) {                                   // A: 128 x float4
      int r = tid >> 3, c4 = (tid & 7) * 4;
      float4 w = *(const float4*)&X[(size_t)(rowbase + r) * Dc + kk0 + c4];
      *(u16x4*)&As[r][c4] = pack4(w);
    }
#pragma unroll
    for (int it = 0; it < 4; ++it) {                   // B: 1024 x float4
      int i = tid + it * 256;
      int kk = i >> 5, n4 = (i & 31) * 4;
      float4 w = *(const float4*)&W[(size_t)(kk0 + kk) * Dc + col0 + n4];
      Bs[n4 + 0][kk] = f2bf(w.x);
      Bs[n4 + 1][kk] = f2bf(w.y);
      Bs[n4 + 2][kk] = f2bf(w.z);
      Bs[n4 + 3][kk] = f2bf(w.w);
    }
    if (kt + 1 < Dc / 32)
      __builtin_prefetch(&W[(size_t)(kk0 + 32) * Dc + col0 + (tid & 127)], 0, 1);
    __syncthreads();
    v16bf a = make_frag(&As[lane & 15][kbA], &As[lane & 15][kbA + 8]);
    v16bf bfrag = make_frag(&Bs[wv * 16 + (lane & 15)][kbB],
                            &Bs[wv * 16 + (lane & 15)][kbB + 8]);
    acc = wmma_bf16(a, bfrag, acc);
    __syncthreads();
  }
  const int c  = col0 + wv * 16 + (lane & 15);
  const float bval = bias[c];
  const int h = c >> 6, hd = c & 63;
#pragma unroll
  for (int r = 0; r < 8; ++r) {
    int m  = rowbase + r + ((lane >> 4) << 3);
    int bb = m >> 11, s = m & 2047;
    Out[(((size_t)(bb * Hc + h)) * Sc + s) * HDc + hd] = f2bf(acc[r] + bval);
  }
}

// =====================================================================
// Kernel 2: QK^T logits (scaled, causal-masked) -> attn region (fp32).
//   grid = (S/16, H, B), block = 256 (8 waves)
//   Async path: each wave double-buffers its own 16x64 K tile in LDS.
// =====================================================================
__global__ __launch_bounds__(256) void qk_logits_kernel(
    const unsigned short* __restrict__ Qh, const unsigned short* __restrict__ Kh,
    float* __restrict__ attn) {
  const int bz = blockIdx.z, h = blockIdx.y;
  const int qbase = blockIdx.x * 16;
  const unsigned short* Qp = Qh + ((size_t)(bz * Hc + h)) * Sc * HDc;
  const unsigned short* Kp = Kh + ((size_t)(bz * Hc + h)) * Sc * HDc;
  float* Ap = attn + (((size_t)(bz * Hc + h)) * Sc + qbase) * Sc;

  __shared__ unsigned short Qs[16 * 64];
  const int tid = threadIdx.x, lane = tid & 31, wv = tid >> 5;
  if (tid < 128)
    *(u16x8*)&Qs[tid * 8] = *(const u16x8*)(Qp + (size_t)qbase * HDc + tid * 8);
  __syncthreads();

  const int kbA = (lane >> 4) * 16, kbB = (lane >> 4) * 16;

#if USE_ASYNC_LDS
  __shared__ unsigned short Kbuf[8][2][16 * 64];       // per-wave double buffer
  auto issue = [&](int step, int buf) {
    const unsigned short* src = Kp + (size_t)(step * 128 + wv * 16) * HDc;
    unsigned short* dst = &Kbuf[wv][buf][0];
#pragma unroll
    for (int c = 0; c < 4; ++c)
      async_copy16(src + c * 256 + lane * 8, dst + c * 256 + lane * 8);
  };
  issue(0, 0);
  int buf = 0;
  for (int step = 0; step < 16; ++step) {
    if (step + 1 < 16) {
      issue(step + 1, buf ^ 1);
      __builtin_amdgcn_s_wait_asynccnt(4);             // retire step's 4 DMAs
    } else {
      __builtin_amdgcn_s_wait_asynccnt(0);
    }
    const int ktile = step * 8 + wv;
    v8f acc = {};
#pragma unroll
    for (int kk = 0; kk < HDc; kk += 32) {
      v16bf a = make_frag(&Qs[(lane & 15) * 64 + kk + kbA],
                          &Qs[(lane & 15) * 64 + kk + kbA + 8]);
      const unsigned short* kr = &Kbuf[wv][buf][(lane & 15) * 64 + kk];
      v16bf bfrag = make_frag(kr + kbB, kr + kbB + 8);
      acc = wmma_bf16(a, bfrag, acc);
    }
    const int col = ktile * 16 + (lane & 15);
#pragma unroll
    for (int r = 0; r < 8; ++r) {
      int qr = r + ((lane >> 4) << 3);
      float v = acc[r] * 0.125f;                       // 1/sqrt(64)
      if (col > qbase + qr) v = NEGV;                  // causal mask
      Ap[(size_t)qr * Sc + col] = v;
    }
    buf ^= 1;
  }
#else
  for (int ktile = wv; ktile < Sc / 16; ktile += 8) {
    v8f acc = {};
#pragma unroll
    for (int kk = 0; kk < HDc; kk += 32) {
      v16bf a = make_frag(&Qs[(lane & 15) * 64 + kk + kbA],
                          &Qs[(lane & 15) * 64 + kk + kbA + 8]);
      const unsigned short* kr = Kp + (size_t)(ktile * 16 + (lane & 15)) * HDc + kk;
      v16bf bfrag = make_frag(kr + kbB, kr + kbB + 8);
      acc = wmma_bf16(a, bfrag, acc);
    }
    const int col = ktile * 16 + (lane & 15);
#pragma unroll
    for (int r = 0; r < 8; ++r) {
      int qr = r + ((lane >> 4) << 3);
      float v = acc[r] * 0.125f;
      if (col > qbase + qr) v = NEGV;
      Ap[(size_t)qr * Sc + col] = v;
    }
  }
#endif
}

// =====================================================================
// Kernel 3: row softmax in place.  grid = B*H*S blocks, block = 256
// =====================================================================
__global__ __launch_bounds__(256) void softmax_row_kernel(float* __restrict__ attn) {
  const size_t row = blockIdx.x;
  float* p = attn + row * (size_t)Sc;
  const float4* p4 = (const float4*)p;
  const int tid = threadIdx.x;
  float4 a = p4[2 * tid], b = p4[2 * tid + 1];
  float m = fmaxf(fmaxf(fmaxf(a.x, a.y), fmaxf(a.z, a.w)),
                  fmaxf(fmaxf(b.x, b.y), fmaxf(b.z, b.w)));
  __shared__ float red[256];
  red[tid] = m; __syncthreads();
  for (int s = 128; s > 0; s >>= 1) {
    if (tid < s) red[tid] = fmaxf(red[tid], red[tid + s]);
    __syncthreads();
  }
  m = red[0]; __syncthreads();
  a.x = __expf(a.x - m); a.y = __expf(a.y - m);
  a.z = __expf(a.z - m); a.w = __expf(a.w - m);
  b.x = __expf(b.x - m); b.y = __expf(b.y - m);
  b.z = __expf(b.z - m); b.w = __expf(b.w - m);
  float sum = (a.x + a.y + a.z + a.w) + (b.x + b.y + b.z + b.w);
  red[tid] = sum; __syncthreads();
  for (int s = 128; s > 0; s >>= 1) {
    if (tid < s) red[tid] += red[tid + s];
    __syncthreads();
  }
  const float inv = 1.0f / red[0];
  a.x *= inv; a.y *= inv; a.z *= inv; a.w *= inv;
  b.x *= inv; b.y *= inv; b.z *= inv; b.w *= inv;
  float4* q4 = (float4*)p;
  q4[2 * tid] = a; q4[2 * tid + 1] = b;
}

// =====================================================================
// Kernel 4: out_h = attn @ V   grid = (S/32, H, B), block = 256
//   V tile (32x64 bf16, contiguous 4KB) async-DMAed, double-buffered.
// =====================================================================
__global__ __launch_bounds__(256) void av_kernel(
    const float* __restrict__ attn, const unsigned short* __restrict__ Vh,
    unsigned short* __restrict__ Oh) {
  const int bz = blockIdx.z, h = blockIdx.y;
  const int qbase = blockIdx.x * 32;
  const float* Ap = attn + (((size_t)(bz * Hc + h)) * Sc + qbase) * Sc;
  const unsigned short* Vp = Vh + ((size_t)(bz * Hc + h)) * Sc * HDc;
  unsigned short* Op = Oh + ((size_t)(bz * Hc + h)) * Sc * HDc;

  __shared__ unsigned short As2[32][32];    // attn tile [q][k] bf16
  __shared__ unsigned short Vs[2][32 * 64]; // V tile [k][hd] (DMA layout)

  const int tid = threadIdx.x, lane = tid & 31, wv = tid >> 5;
  const int qsub = wv >> 2, ntile = wv & 3;
  const int kbA = (lane >> 4) * 16, kbB = (lane >> 4) * 16;
  const int hd = ntile * 16 + (lane & 15);

  v8f acc = {};
#if USE_ASYNC_LDS
  async_copy16(Vp + tid * 8, &Vs[0][tid * 8]);         // tile 0
  int buf = 0;
#endif
  for (int kt = 0; kt < Sc / 32; ++kt) {
    const int kk0 = kt * 32;
    {                                                  // attn tile: 256 x float4
      int r = tid >> 3, c4 = (tid & 7) * 4;
      float4 w = *(const float4*)&Ap[(size_t)r * Sc + kk0 + c4];
      *(u16x4*)&As2[r][c4] = pack4(w);
    }
#if USE_ASYNC_LDS
    if (kt + 1 < Sc / 32) {
      async_copy16(Vp + (size_t)(kk0 + 32) * HDc + tid * 8, &Vs[buf ^ 1][tid * 8]);
      __builtin_amdgcn_s_wait_asynccnt(1);
    } else {
      __builtin_amdgcn_s_wait_asynccnt(0);
    }
    const unsigned short* vtile = &Vs[buf][0];
#else
    *(u16x8*)&Vs[0][tid * 8] = *(const u16x8*)(Vp + (size_t)kk0 * HDc + tid * 8);
    const unsigned short* vtile = &Vs[0][0];
#endif
    __syncthreads();
    v16bf a = make_frag(&As2[qsub * 16 + (lane & 15)][kbA],
                        &As2[qsub * 16 + (lane & 15)][kbA + 8]);
    FragCvt bf;
#pragma unroll
    for (int i = 0; i < 16; ++i)                       // gather column hd
      bf.u[i] = vtile[(kbB + i) * 64 + hd];
    acc = wmma_bf16(a, bf.b, acc);
    __syncthreads();
#if USE_ASYNC_LDS
    buf ^= 1;
#endif
  }
#pragma unroll
  for (int r = 0; r < 8; ++r) {
    int q = qbase + qsub * 16 + r + ((lane >> 4) << 3);
    Op[(size_t)q * HDc + hd] = f2bf(acc[r]);
  }
}

// =====================================================================
// Kernel 5: o = merge_heads(out_h) @ Wo + bo   grid = (M/16, D/128)
// =====================================================================
__global__ __launch_bounds__(256) void out_proj_kernel(
    const unsigned short* __restrict__ Oh, const float* __restrict__ Wo,
    const float* __restrict__ bo, float* __restrict__ o) {
  __shared__ unsigned short As[16][32];
  __shared__ unsigned short Bs[128][32];
  const int tid = threadIdx.x, lane = tid & 31, wv = tid >> 5;
  const int rowbase = blockIdx.x * 16;
  const int col0    = blockIdx.y * 128;
  const int kbA = (lane >> 4) * 16, kbB = (lane >> 4) * 16;

  v8f acc = {};
  for (int kt = 0; kt < Dc / 32; ++kt) {
    const int kk0 = kt * 32;
    if (tid < 64) {                       // A: 64 x u16x8 (bf16, head-gather)
      int r = tid >> 2, c8 = (tid & 3) * 8;
      int m = rowbase + r, bb = m >> 11, s = m & 2047;
      int h = kk0 >> 6, hd0 = (kk0 & 63) + c8;   // same h across the 32-wide tile
      *(u16x8*)&As[r][c8] =
          *(const u16x8*)(Oh + (((size_t)(bb * Hc + h)) * Sc + s) * HDc + hd0);
    }
#pragma unroll
    for (int it = 0; it < 4; ++it) {      // B: 1024 x float4
      int i = tid + it * 256;
      int kk = i >> 5, n4 = (i & 31) * 4;
      float4 w = *(const float4*)&Wo[(size_t)(kk0 + kk) * Dc + col0 + n4];
      Bs[n4 + 0][kk] = f2bf(w.x);
      Bs[n4 + 1][kk] = f2bf(w.y);
      Bs[n4 + 2][kk] = f2bf(w.z);
      Bs[n4 + 3][kk] = f2bf(w.w);
    }
    if (kt + 1 < Dc / 32)
      __builtin_prefetch(&Wo[(size_t)(kk0 + 32) * Dc + col0 + (tid & 127)], 0, 1);
    __syncthreads();
    v16bf a = make_frag(&As[lane & 15][kbA], &As[lane & 15][kbA + 8]);
    v16bf bfrag = make_frag(&Bs[wv * 16 + (lane & 15)][kbB],
                            &Bs[wv * 16 + (lane & 15)][kbB + 8]);
    acc = wmma_bf16(a, bfrag, acc);
    __syncthreads();
  }
  const int c = col0 + wv * 16 + (lane & 15);
  const float bval = bo[c];
#pragma unroll
  for (int r = 0; r < 8; ++r) {
    int m = rowbase + r + ((lane >> 4) << 3);
    o[(size_t)m * Dc + c] = acc[r] + bval;
  }
}

// =====================================================================
extern "C" void kernel_launch(void* const* d_in, const int* in_sizes, int n_in,
                              void* d_out, int out_size, void* d_ws, size_t ws_size,
                              hipStream_t stream) {
  // input order: k, q, v, mask, Wq, bq, Wk, bk, Wv, bv, Wo, bo
  const float* k_in = (const float*)d_in[0];
  const float* q_in = (const float*)d_in[1];
  const float* v_in = (const float*)d_in[2];
  // d_in[3] = mask (causal tril) -- applied analytically from indices
  const float* Wq = (const float*)d_in[4];
  const float* bq = (const float*)d_in[5];
  const float* Wk = (const float*)d_in[6];
  const float* bk = (const float*)d_in[7];
  const float* Wv = (const float*)d_in[8];
  const float* bv = (const float*)d_in[9];
  const float* Wo = (const float*)d_in[10];
  const float* bo = (const float*)d_in[11];

  float* o    = (float*)d_out;                       // [B,S,D]
  float* attn = o + (size_t)Bc * Sc * Dc;            // [B,H,S,S]

  const size_t HSZ = (size_t)Bc * Hc * Sc * HDc;     // 4,194,304 elems
  unsigned short* Qh = (unsigned short*)d_ws;
  unsigned short* Kh = Qh + HSZ;
  unsigned short* Vh = Kh + HSZ;
  unsigned short* Oh = Vh + HSZ;

  const int M = Bc * Sc;                             // 4096

  qkv_proj_kernel<<<dim3(M / 16, Dc / 128, 3), 256, 0, stream>>>(
      k_in, q_in, v_in, Wq, bq, Wk, bk, Wv, bv, Qh, Kh, Vh);

  qk_logits_kernel<<<dim3(Sc / 16, Hc, Bc), 256, 0, stream>>>(Qh, Kh, attn);

  softmax_row_kernel<<<dim3(Bc * Hc * Sc), 256, 0, stream>>>(attn);

  av_kernel<<<dim3(Sc / 32, Hc, Bc), 256, 0, stream>>>(attn, Vh, Oh);

  out_proj_kernel<<<dim3(M / 16, Dc / 128, 1), 256, 0, stream>>>(Oh, Wo, bo, o);
}
